// MultiHeadAttention_1683627180863
// MI455X (gfx1250) — matrix operand order repaired and checked by reading, not compiled, not run on hardware
//
#include <hip/hip_runtime.h>

// ---------------------------------------------------------------------------
// MHA block for MI455X (gfx1250, wave32, WMMA 16x16x32 bf16, f32 accumulate)
//
// Pipeline (all on `stream`):
//   1) k_transpose      : Wq/Wk/Wv/Wo  f32[K,N] -> bf16 Wt[N,K]   (LDS tiled)
//   2) k_proj<MODE> x3  : X @ W + b -> q,k as bf16 [b,h,s,d]; v as bf16 [b,h,d,s]
//   3) k_scores         : (q kT)*1/8 -> attn raw f32 (written straight to d_out)
//   4) k_softmax        : row softmax in place over attn (mask applied)
//   5) k_context        : attn @ v -> ctx f32 [b,s, h*dv]
//   6) k_outproj        : ctx @ Wo + bo + residual(Q) -> y f32
//   7) k_layernorm      : LN(y)*gamma+beta -> d_out[:B*S*D]
// ---------------------------------------------------------------------------

#define B_   2
#define S_   2048
#define H_   16
#define DK_  64
#define DM_  1024
#define BH_  (B_ * H_)          // 32
#define M_   (B_ * S_)          // 4096
#define LN_EPS 1e-5f

typedef __attribute__((ext_vector_type(16))) __bf16       v16bf;
typedef __attribute__((ext_vector_type(8)))  float        v8f;
typedef __attribute__((ext_vector_type(4)))  float        f4;
typedef __attribute__((ext_vector_type(4)))  unsigned int u4;

union Frag16 { v16bf v; u4 u[2]; };

// ---- WMMA wrapper ----------------------------------------------------------
__device__ __forceinline__ v8f wmma_bf16(v16bf a, v16bf b, v8f c) {
  // (neg_a, A, neg_b, B, c_mod, C, reuse_a, reuse_b)
  return __builtin_amdgcn_wmma_f32_16x16x32_bf16(false, a, false, b, (short)0, c,
                                                 false, false);
}

// ---- Fragment loaders (layouts per CDNA5 ISA 7.12.2, wave32) ---------------
// A 16x32 bf16: lane<16 -> M=lane, K = {0..7, 16..23}; lane>=16 -> K = {8..15, 24..31}
__device__ __forceinline__ v16bf load_a_bf16(const __bf16* __restrict__ base,
                                             int lda, int m0, int k0, int lane) {
  int m  = m0 + (lane & 15);
  int kb = k0 + ((lane >> 4) << 3);
  const u4* p0 = reinterpret_cast<const u4*>(base + (size_t)m * lda + kb);
  const u4* p1 = reinterpret_cast<const u4*>(base + (size_t)m * lda + kb + 16);
  Frag16 f; f.u[0] = *p0; f.u[1] = *p1;
  return f.v;
}

__device__ __forceinline__ v16bf load_a_f32(const float* __restrict__ base,
                                            int lda, int m0, int k0, int lane) {
  int m  = m0 + (lane & 15);
  int kb = k0 + ((lane >> 4) << 3);
  const f4* p0 = reinterpret_cast<const f4*>(base + (size_t)m * lda + kb);
  const f4* p1 = reinterpret_cast<const f4*>(base + (size_t)m * lda + kb + 16);
  f4 x0 = p0[0], x1 = p0[1], x2 = p1[0], x3 = p1[1];
  v16bf a;
#pragma unroll
  for (int i = 0; i < 4; ++i) {
    a[i]      = (__bf16)x0[i];
    a[4 + i]  = (__bf16)x1[i];
    a[8 + i]  = (__bf16)x2[i];
    a[12 + i] = (__bf16)x3[i];
  }
  return a;
}

// B 32x16 bf16 from a K-contiguous [N,K] source: lane -> N = n0+(lane&15),
// K = k0 + 8*(lane>>4) + {0..7, 16..23}, two 16B loads per lane.
__device__ __forceinline__ v16bf load_bt_bf16(const __bf16* __restrict__ baseNK,
                                              int ldk, int n0, int k0, int lane) {
  int n  = n0 + (lane & 15);
  int kb = k0 + ((lane >> 4) << 3);
  const u4* p = reinterpret_cast<const u4*>(baseNK + (size_t)n * ldk + kb);
  Frag16 f; f.u[0] = p[0]; f.u[1] = p[2];
  return f.v;
}

// ---------------------------------------------------------------------------
// 1) Weight transpose + bf16 convert: Wt[n*1024 + k] = (bf16)W[k*1024 + n]
// ---------------------------------------------------------------------------
__global__ void k_transpose(const float* __restrict__ W, __bf16* __restrict__ Wt) {
  __shared__ float tile[64][65];
  int n0 = blockIdx.x * 64;     // column block of W (output rows)
  int k0 = blockIdx.y * 64;     // row block of W (output cols)
  for (int i = threadIdx.x; i < 64 * 64; i += 256) {
    int r = i >> 6, c = i & 63;
    tile[r][c] = W[(size_t)(k0 + r) * DM_ + n0 + c];
  }
  __syncthreads();
  for (int i = threadIdx.x; i < 64 * 64; i += 256) {
    int r = i >> 6, c = i & 63;                        // r: n-local, c: k-local
    Wt[(size_t)(n0 + r) * DM_ + k0 + c] = (__bf16)tile[c][r];
  }
}

// ---------------------------------------------------------------------------
// 2) Projection GEMM: Out = X[4096,1024] @ W + b, written bf16.
//    MODE 0: Out[((b*16+h)*2048+s)*64+d]   (q, k)
//    MODE 1: Out[((b*16+h)*64+d)*2048+s]   (v transposed for context GEMM)
// ---------------------------------------------------------------------------
template <int MODE>
__global__ void k_proj(const float* __restrict__ X, const __bf16* __restrict__ Wt,
                       const float* __restrict__ bias, __bf16* __restrict__ Out) {
  int lane = threadIdx.x & 31;
  int wave = threadIdx.x >> 5;
  int n0 = blockIdx.x * 64;
  int m0 = blockIdx.y * 64 + wave * 16;

  v8f acc[4] = {};
  for (int k0 = 0; k0 < DM_; k0 += 64) {   // K unrolled x2: both halves accumulate into acc[s]
    __builtin_prefetch(Wt + (size_t)(n0 + (lane & 15)) * DM_ + k0 + 64, 0, 3);
    v16bf a0 = load_a_f32(X, DM_, m0, k0, lane);
    v16bf a1 = load_a_f32(X, DM_, m0, k0 + 32, lane);
#pragma unroll
    for (int s = 0; s < 4; ++s) {
      v16bf b = load_bt_bf16(Wt, DM_, n0 + s * 16, k0, lane);
      acc[s] = wmma_bf16(a0, b, acc[s]);
    }
#pragma unroll
    for (int s = 0; s < 4; ++s) {
      v16bf b = load_bt_bf16(Wt, DM_, n0 + s * 16, k0 + 32, lane);
      acc[s] = wmma_bf16(a1, b, acc[s]);
    }
  }

  int mhalf = (lane >> 4) << 3;
#pragma unroll
  for (int s = 0; s < 4; ++s) {
    int n = n0 + s * 16 + (lane & 15);
    int h = n >> 6, d = n & 63;
    float bv = bias[n];
#pragma unroll
    for (int r = 0; r < 8; ++r) {
      int m = m0 + mhalf + r;
      int b = m >> 11, seq = m & 2047;
      float val = acc[s][r] + bv;
      size_t idx = (MODE == 1)
        ? ((size_t)((b * H_ + h) * DK_ + d) << 11) + seq
        : ((size_t)(b * H_ + h) << 17) + (size_t)seq * DK_ + d;
      Out[idx] = (__bf16)val;
    }
  }
}

// ---------------------------------------------------------------------------
// 3) Scores: attn_raw[bh, sq, sk] = (q . k) / 8
// ---------------------------------------------------------------------------
__global__ void k_scores(const __bf16* __restrict__ qp, const __bf16* __restrict__ kp,
                         float* __restrict__ attn) {
  int lane = threadIdx.x & 31;
  int wave = threadIdx.x >> 5;
  int bh = blockIdx.z;
  int n0 = blockIdx.x * 64;                 // s_k
  int m0 = blockIdx.y * 64 + wave * 16;     // s_q

  const __bf16* qb = qp + ((size_t)bh << 17);
  const __bf16* kb = kp + ((size_t)bh << 17);

  v8f acc[4] = {};
  v16bf a0 = load_a_bf16(qb, DK_, m0, 0, lane);
  v16bf a1 = load_a_bf16(qb, DK_, m0, 32, lane);
#pragma unroll
  for (int s = 0; s < 4; ++s) {
    v16bf b = load_bt_bf16(kb, DK_, n0 + s * 16, 0, lane);
    acc[s] = wmma_bf16(a0, b, acc[s]);
  }
#pragma unroll
  for (int s = 0; s < 4; ++s) {
    v16bf b = load_bt_bf16(kb, DK_, n0 + s * 16, 32, lane);
    acc[s] = wmma_bf16(a1, b, acc[s]);
  }

  int mhalf = (lane >> 4) << 3;
#pragma unroll
  for (int s = 0; s < 4; ++s) {
    int n = n0 + s * 16 + (lane & 15);
#pragma unroll
    for (int r = 0; r < 8; ++r) {
      int m = m0 + mhalf + r;
      attn[((size_t)bh << 22) + ((size_t)m << 11) + n] = acc[s][r] * 0.125f;
    }
  }
}

// ---------------------------------------------------------------------------
// 4) Row softmax in place over attn; mask[b, sq, sk] -> -1e9
// ---------------------------------------------------------------------------
__global__ void k_softmax(float* __restrict__ attn,
                          const unsigned char* __restrict__ mask) {
  int row = blockIdx.x;              // row = bh*2048 + sq
  int sq  = row & 2047;
  int b   = row >> 15;               // (row>>11)>>4
  float* p = attn + ((size_t)row << 11);
  const unsigned char* mp = mask + (((size_t)b << 11) + sq) * S_;
  int t = threadIdx.x;

  float v[8];
  float mx = -1e30f;
#pragma unroll
  for (int i = 0; i < 8; ++i) {
    int c = t * 8 + i;
    float x = p[c];
    if (mp[c]) x = -1e9f;
    v[i] = x;
    mx = fmaxf(mx, x);
  }
  __shared__ float red[256];
  red[t] = mx; __syncthreads();
  for (int s = 128; s > 0; s >>= 1) {
    if (t < s) red[t] = fmaxf(red[t], red[t + s]);
    __syncthreads();
  }
  mx = red[0]; __syncthreads();

  float sum = 0.f;
#pragma unroll
  for (int i = 0; i < 8; ++i) { v[i] = __expf(v[i] - mx); sum += v[i]; }
  red[t] = sum; __syncthreads();
  for (int s = 128; s > 0; s >>= 1) {
    if (t < s) red[t] += red[t + s];
    __syncthreads();
  }
  float inv = 1.0f / red[0];
#pragma unroll
  for (int i = 0; i < 8; ++i) p[t * 8 + i] = v[i] * inv;
}

// ---------------------------------------------------------------------------
// 5) Context: ctx[b, s, h*64+d] = attn[bh, s, :] @ v[bh, :, d]   (vT layout)
// ---------------------------------------------------------------------------
__global__ void k_context(const float* __restrict__ attn,
                          const __bf16* __restrict__ vtp,
                          float* __restrict__ ctx) {
  int lane = threadIdx.x & 31;
  int wave = threadIdx.x >> 5;
  int bh = blockIdx.y;
  int m0 = blockIdx.x * 64 + wave * 16;
  int b = bh >> 4, h = bh & 15;

  const float*  ab = attn + ((size_t)bh << 22);
  const __bf16* vb = vtp + ((size_t)bh << 17);   // [d, s] ldk = 2048

  v8f acc[4] = {};
  for (int k0 = 0; k0 < S_; k0 += 64) {    // K unrolled x2
    v16bf a0 = load_a_f32(ab, S_, m0, k0, lane);
    v16bf a1 = load_a_f32(ab, S_, m0, k0 + 32, lane);
#pragma unroll
    for (int s = 0; s < 4; ++s) {
      v16bf bfr = load_bt_bf16(vb, S_, s * 16, k0, lane);
      acc[s] = wmma_bf16(a0, bfr, acc[s]);
    }
#pragma unroll
    for (int s = 0; s < 4; ++s) {
      v16bf bfr = load_bt_bf16(vb, S_, s * 16, k0 + 32, lane);
      acc[s] = wmma_bf16(a1, bfr, acc[s]);
    }
  }

  int mhalf = (lane >> 4) << 3;
#pragma unroll
  for (int s = 0; s < 4; ++s) {
    int d = s * 16 + (lane & 15);
#pragma unroll
    for (int r = 0; r < 8; ++r) {
      int m = m0 + mhalf + r;
      ctx[((size_t)(b * S_ + m) << 10) + h * DK_ + d] = acc[s][r];
    }
  }
}

// ---------------------------------------------------------------------------
// 6) Output projection + bias + residual: y = ctx @ Wo + bo + Q
// ---------------------------------------------------------------------------
__global__ void k_outproj(const float* __restrict__ ctx, const __bf16* __restrict__ Wt,
                          const float* __restrict__ bo, const float* __restrict__ resid,
                          float* __restrict__ y) {
  int lane = threadIdx.x & 31;
  int wave = threadIdx.x >> 5;
  int n0 = blockIdx.x * 64;
  int m0 = blockIdx.y * 64 + wave * 16;

  v8f acc[4] = {};
  for (int k0 = 0; k0 < DM_; k0 += 64) {   // K unrolled x2
    __builtin_prefetch(Wt + (size_t)(n0 + (lane & 15)) * DM_ + k0 + 64, 0, 3);
    v16bf a0 = load_a_f32(ctx, DM_, m0, k0, lane);
    v16bf a1 = load_a_f32(ctx, DM_, m0, k0 + 32, lane);
#pragma unroll
    for (int s = 0; s < 4; ++s) {
      v16bf b = load_bt_bf16(Wt, DM_, n0 + s * 16, k0, lane);
      acc[s] = wmma_bf16(a0, b, acc[s]);
    }
#pragma unroll
    for (int s = 0; s < 4; ++s) {
      v16bf b = load_bt_bf16(Wt, DM_, n0 + s * 16, k0 + 32, lane);
      acc[s] = wmma_bf16(a1, b, acc[s]);
    }
  }

  int mhalf = (lane >> 4) << 3;
#pragma unroll
  for (int s = 0; s < 4; ++s) {
    int n = n0 + s * 16 + (lane & 15);
    float bv = bo[n];
#pragma unroll
    for (int r = 0; r < 8; ++r) {
      int m = m0 + mhalf + r;
      size_t idx = ((size_t)m << 10) + n;
      y[idx] = acc[s][r] + bv + resid[idx];
    }
  }
}

// ---------------------------------------------------------------------------
// 7) LayerNorm over last dim (1024) -> d_out
// ---------------------------------------------------------------------------
__global__ void k_layernorm(const float* __restrict__ y,
                            const float* __restrict__ gamma,
                            const float* __restrict__ beta,
                            float* __restrict__ out) {
  int row = blockIdx.x;
  const float* p = y + ((size_t)row << 10);
  int t = threadIdx.x;

  float v[4];
  float s = 0.f, s2 = 0.f;
#pragma unroll
  for (int i = 0; i < 4; ++i) {
    v[i] = p[t * 4 + i];
    s += v[i];
    s2 += v[i] * v[i];
  }
  __shared__ float rs[256], rs2[256];
  rs[t] = s; rs2[t] = s2; __syncthreads();
  for (int k = 128; k > 0; k >>= 1) {
    if (t < k) { rs[t] += rs[t + k]; rs2[t] += rs2[t + k]; }
    __syncthreads();
  }
  float mu  = rs[0] * (1.0f / DM_);
  float var = rs2[0] * (1.0f / DM_) - mu * mu;
  float inv = rsqrtf(var + LN_EPS);
#pragma unroll
  for (int i = 0; i < 4; ++i) {
    int c = t * 4 + i;
    out[((size_t)row << 10) + c] = (v[i] - mu) * inv * gamma[c] + beta[c];
  }
}

// ---------------------------------------------------------------------------
extern "C" void kernel_launch(void* const* d_in, const int* in_sizes, int n_in,
                              void* d_out, int out_size, void* d_ws, size_t ws_size,
                              hipStream_t stream) {
  const float* Qin  = (const float*)d_in[0];
  const float* Kin  = (const float*)d_in[1];
  const float* Vin  = (const float*)d_in[2];
  const unsigned char* mask = (const unsigned char*)d_in[3];
  const float* Wq = (const float*)d_in[4];  const float* bq = (const float*)d_in[5];
  const float* Wk = (const float*)d_in[6];  const float* bk = (const float*)d_in[7];
  const float* Wv = (const float*)d_in[8];  const float* bv = (const float*)d_in[9];
  const float* Wo = (const float*)d_in[10]; const float* bo = (const float*)d_in[11];
  const float* gamma = (const float*)d_in[12];
  const float* beta  = (const float*)d_in[13];

  float* out_ln = (float*)d_out;                               // [B,S,DM]
  float* attn   = (float*)d_out + (size_t)M_ * DM_;            // [B,H,S,S]

  char* ws = (char*)d_ws;
  __bf16* wtq = (__bf16*)(ws + 0);                 // 2 MB each, bf16 [N,K]
  __bf16* wtk = (__bf16*)(ws + (2u << 20));
  __bf16* wtv = (__bf16*)(ws + (4u << 20));
  __bf16* wto = (__bf16*)(ws + (6u << 20));
  __bf16* qp  = (__bf16*)(ws + (8u << 20));        // 8 MB each
  __bf16* kp  = (__bf16*)(ws + (16u << 20));
  __bf16* vtp = (__bf16*)(ws + (24u << 20));
  float*  ctx = (float*)(ws + (32u << 20));        // 16 MB
  float*  yb  = (float*)(ws + (48u << 20));        // 16 MB

  dim3 bt(256), gt(16, 16);
  k_transpose<<<gt, bt, 0, stream>>>(Wq, wtq);
  k_transpose<<<gt, bt, 0, stream>>>(Wk, wtk);
  k_transpose<<<gt, bt, 0, stream>>>(Wv, wtv);
  k_transpose<<<gt, bt, 0, stream>>>(Wo, wto);

  dim3 bg(128), gp(16, 64);
  k_proj<0><<<gp, bg, 0, stream>>>(Qin, wtq, bq, qp);
  k_proj<0><<<gp, bg, 0, stream>>>(Kin, wtk, bk, kp);
  k_proj<1><<<gp, bg, 0, stream>>>(Vin, wtv, bv, vtp);

  k_scores<<<dim3(32, 32, BH_), bg, 0, stream>>>(qp, kp, attn);
  k_softmax<<<dim3(BH_ * S_), dim3(256), 0, stream>>>(attn, mask);
  k_context<<<dim3(32, BH_), bg, 0, stream>>>(attn, vtp, ctx);
  k_outproj<<<gp, bg, 0, stream>>>(ctx, wto, bo, Qin, yb);
  k_layernorm<<<dim3(M_), dim3(256), 0, stream>>>(yb, gamma, beta, out_ln);
}
